// SOLOv2_41850161332608
// MI455X (gfx1250) — 8-bit matrix operand order repaired and checked by reading, compiled, not run on hardware
//
#include <hip/hip_runtime.h>
#include <hip/hip_bf16.h>

#define N_MASK   512
#define MASK_HW  61440
#define MASK_DW  15360
#define HEAT_TOT 9830400
#define HEAT_W   320
#define HEAT_H   192
#define SIGMA    2.0f

typedef __attribute__((ext_vector_type(16))) int   v16i;
typedef __attribute__((ext_vector_type(8)))  float v8f;
typedef __attribute__((ext_vector_type(4)))  int   v4i;
typedef __attribute__((ext_vector_type(2)))  int   v2i;

__global__ __launch_bounds__(N_MASK) void k_argsort(const float* __restrict__ s,
                                                    int* __restrict__ order) {
  __shared__ float sh[N_MASK];
  int i = threadIdx.x;
  sh[i] = s[i];
  __syncthreads();
  float si = sh[i];
  int rank = 0;
  for (int j = 0; j < N_MASK; ++j) {
    float sj = sh[j];
    rank += (sj > si) || (sj == si && j < i);
  }
  order[rank] = i;
}

__global__ __launch_bounds__(256) void k_quantize(const float* __restrict__ masks,
                                                  const int* __restrict__ order,
                                                  unsigned int* __restrict__ fmq) {
  int row = blockIdx.y;
  int idx = blockIdx.x * 256 + threadIdx.x;
  int src = order[row];
  const float4 m = *(const float4*)(masks + (size_t)src * MASK_HW + (size_t)idx * 4);
  unsigned int w = 0;
  w |= (m.x != 0.0f) ? 0x00000038u : 0u;
  w |= (m.y != 0.0f) ? 0x00003800u : 0u;
  w |= (m.z != 0.0f) ? 0x00380000u : 0u;
  w |= (m.w != 0.0f) ? 0x38000000u : 0u;
  fmq[(size_t)row * MASK_DW + idx] = w;
}

__device__ __forceinline__ void load_fragA(v16i& a, const unsigned char* rowA,
                                           int kb, int hi) {
  #pragma unroll
  for (int p = 0; p < 8; ++p) {
    int off = ((p >> 1) * 32) + hi * 16 + ((p & 1) * 8);
    v2i q = *(const v2i*)(rowA + kb + off);
    a[2 * p]     = q.x;
    a[2 * p + 1] = q.y;
  }
}

__device__ __forceinline__ void load_fragB(v16i& bm, const unsigned char* rowB,
                                           int kb, int hi) {
  #pragma unroll
  for (int g = 0; g < 4; ++g) {
    int off = g * 32 + hi * 16;
    v4i q = *(const v4i*)(rowB + kb + off);
    bm[4 * g]     = q.x;
    bm[4 * g + 1] = q.y;
    bm[4 * g + 2] = q.z;
    bm[4 * g + 3] = q.w;
  }
}

__global__ __launch_bounds__(32) void k_gram_wmma(const unsigned char* __restrict__ fmq,
                                                  float* __restrict__ inter) {
  int b = blockIdx.x;
  int ti = 0, rem = b;
  while (rem >= (32 - ti)) { rem -= (32 - ti); ++ti; }
  int tj = ti + rem;

  const int lane = threadIdx.x;
  const int hi   = lane >> 4;
  const int l    = lane & 15;

  const unsigned char* rowA = fmq + (size_t)(ti * 16 + l) * MASK_HW;
  const unsigned char* rowB = fmq + (size_t)(tj * 16 + l) * MASK_HW;

  v16i a0, b0, a1, b1;
  load_fragA(a0, rowA, 0, hi);
  load_fragB(b0, rowB, 0, hi);

  v8f c = {};
  #pragma unroll 1
  for (int kb = 0; kb < MASK_HW - 256; kb += 256) {
    __builtin_prefetch(rowA + kb + 256, 0, 3);
    __builtin_prefetch(rowB + kb + 256, 0, 3);
    load_fragA(a1, rowA, kb + 128, hi);
    load_fragB(b1, rowB, kb + 128, hi);
    c = __builtin_amdgcn_wmma_f32_16x16x128_fp8_fp8(a0, b0, (short)0, c, false, false);
    load_fragA(a0, rowA, kb + 256, hi);
    load_fragB(b0, rowB, kb + 256, hi);
    c = __builtin_amdgcn_wmma_f32_16x16x128_fp8_fp8(a1, b1, (short)0, c, false, false);
  }
  load_fragA(a1, rowA, MASK_HW - 128, hi);
  load_fragB(b1, rowB, MASK_HW - 128, hi);
  c = __builtin_amdgcn_wmma_f32_16x16x128_fp8_fp8(a0, b0, (short)0, c, false, false);
  c = __builtin_amdgcn_wmma_f32_16x16x128_fp8_fp8(a1, b1, (short)0, c, false, false);

  #pragma unroll
  for (int r = 0; r < 8; ++r) {
    inter[(size_t)(ti * 16 + r + 8 * hi) * N_MASK + (tj * 16 + l)] = c[r];
  }
}

__global__ __launch_bounds__(N_MASK) void k_decay(const float* __restrict__ inter,
                                                  const int* __restrict__ order,
                                                  const float* __restrict__ scores,
                                                  const int* __restrict__ labels,
                                                  float* __restrict__ new_scores) {
  __shared__ float area[N_MASK];
  __shared__ int   lab[N_MASK];
  __shared__ float comp[N_MASK];
  int j = threadIdx.x;
  int oj = order[j];
  lab[j]  = labels[oj];
  area[j] = inter[(size_t)j * N_MASK + j];
  __syncthreads();
  int   lj = lab[j];
  float aj = area[j];

  float cmax = 0.0f;
  for (int i = 0; i < j; ++i) {
    if (lab[i] == lj) {
      float in = inter[(size_t)i * N_MASK + j];
      float d  = in / (area[i] + aj - in);
      cmax = fmaxf(cmax, d);
    }
  }
  comp[j] = cmax;
  __syncthreads();

  float coeff = 1.0f;
  for (int i = 0; i < j; ++i) {
    if (lab[i] == lj) {
      float in = inter[(size_t)i * N_MASK + j];
      float d  = in / (area[i] + aj - in);
      float v  = __expf(-SIGMA * (d * d - comp[i] * comp[i]));
      coeff = fminf(coeff, v);
    }
  }
  new_scores[j] = scores[oj] * coeff;
}

__global__ __launch_bounds__(N_MASK) void k_finalize(const float* __restrict__ new_scores,
                                                     const int* __restrict__ order,
                                                     const int* __restrict__ labels,
                                                     int* __restrict__ perm,
                                                     float* __restrict__ out_scores,
                                                     float* __restrict__ out_labels,
                                                     float* __restrict__ out_keep) {
  __shared__ float sh[N_MASK];
  __shared__ int   ord2[N_MASK];
  int i = threadIdx.x;
  sh[i] = new_scores[i];
  __syncthreads();
  float si = sh[i];
  int rank = 0;
  for (int j = 0; j < N_MASK; ++j) {
    float sj = sh[j];
    rank += (sj > si) || (sj == si && j < i);
  }
  ord2[rank] = i;
  __syncthreads();
  int o2 = ord2[i];
  int p  = order[o2];
  perm[i]       = p;
  out_scores[i] = new_scores[o2];
  out_labels[i] = (float)labels[p];
  out_keep[i]   = (float)p;
}

__global__ __launch_bounds__(256) void k_gather_masks(const float* __restrict__ masks,
                                                      const int* __restrict__ perm,
                                                      float* __restrict__ out) {
  int row = blockIdx.y;
  int idx = blockIdx.x * 256 + threadIdx.x;
  int p = perm[row];
  const float4 v = *(const float4*)(masks + (size_t)p * MASK_HW + (size_t)idx * 4);
  *(float4*)(out + (size_t)row * MASK_HW + (size_t)idx * 4) = v;
}

__global__ __launch_bounds__(256) void k_points_nms(const float* __restrict__ heat,
                                                    float* __restrict__ out) {
  int g = blockIdx.x * 256 + threadIdx.x;
  if (g >= HEAT_TOT) return;
  int w = g % HEAT_W;
  int h = (g / HEAT_W) % HEAT_H;
  float v = heat[g];
  float m = v;
  if (h > 0)          m = fmaxf(m, heat[g - HEAT_W]);
  if (w > 0)          m = fmaxf(m, heat[g - 1]);
  if (h > 0 && w > 0) m = fmaxf(m, heat[g - HEAT_W - 1]);
  out[g] = (m == v) ? v : 0.0f;
}

extern "C" void kernel_launch(void* const* d_in, const int* in_sizes, int n_in,
                              void* d_out, int out_size, void* d_ws, size_t ws_size,
                              hipStream_t stream) {
  const float* masks  = (const float*)d_in[0];
  const float* scores = (const float*)d_in[1];
  const float* heat   = (const float*)d_in[2];
  const int*   labels = (const int*)d_in[3];

  char* ws = (char*)d_ws;
  unsigned char* fmq   = (unsigned char*)ws;
  float*         inter = (float*)(ws + (size_t)N_MASK * MASK_HW);
  int*           order = (int*)((char*)inter + (size_t)N_MASK * N_MASK * 4);
  int*           perm  = order + N_MASK;
  float*         nsc   = (float*)(perm + N_MASK);

  float* out_scores = (float*)d_out;
  float* out_labels = out_scores + N_MASK;
  float* out_masks  = out_labels + N_MASK;
  float* out_keep   = out_masks + (size_t)N_MASK * MASK_HW;
  float* out_heat   = out_keep + N_MASK;

  k_argsort<<<1, N_MASK, 0, stream>>>(scores, order);

  k_quantize<<<dim3(MASK_DW / 256, N_MASK), 256, 0, stream>>>(masks, order,
                                                              (unsigned int*)fmq);

  k_gram_wmma<<<528, 32, 0, stream>>>(fmq, inter);

  k_decay<<<1, N_MASK, 0, stream>>>(inter, order, scores, labels, nsc);

  k_finalize<<<1, N_MASK, 0, stream>>>(nsc, order, labels, perm,
                                       out_scores, out_labels, out_keep);

  k_gather_masks<<<dim3(MASK_DW / 256, N_MASK), 256, 0, stream>>>(masks, perm, out_masks);

  k_points_nms<<<(HEAT_TOT + 255) / 256, 256, 0, stream>>>(heat, out_heat);
}
